// MRAAttention_71674414235765
// MI455X (gfx1250) — matrix operand order repaired and checked by reading, not compiled, not run on hardware
//
#include <hip/hip_runtime.h>

// ---------------- problem constants ----------------
#define DM      1024
#define SEQ     4096
#define BATCH   2
#define NHEADS  16
#define HD      64
#define BLK     32
#define NBLK    128            // SEQ/BLK
#define BH      32             // BATCH*NHEADS
#define NUMSEL  1024           // NUM_BLOCK
#define SCALE   0.125f         // 1/sqrt(64)

typedef __attribute__((ext_vector_type(16))) __bf16 v16bf;
typedef __attribute__((ext_vector_type(8)))  __bf16 v8bf;
typedef __attribute__((ext_vector_type(8)))  float  v8f;

__device__ __forceinline__ v8f wmma_bf16(v16bf a, v16bf b, v8f c) {
    return __builtin_amdgcn_wmma_f32_16x16x32_bf16(false, a, false, b, (short)0, c, false, false);
}

__device__ __forceinline__ v16bf pack16(v8bf lo, v8bf hi) {
    v16bf r;
#pragma unroll
    for (int i = 0; i < 8; ++i) { r[i] = lo[i]; r[i + 8] = hi[i]; }
    return r;
}

// async 16B global -> LDS copy (per-lane), tracked by ASYNCcnt
__device__ __forceinline__ void async_copy_b128(void* lds, const void* g) {
    unsigned ldsOff = (unsigned)(size_t)lds;   // low 32 bits of flat LDS addr = wave-relative LDS address
    asm volatile("global_load_async_to_lds_b128 %0, %1, off"
                 :: "v"(ldsOff), "v"(g) : "memory");
}

__device__ __forceinline__ void atomicMaxF(float* addr, float val) {
    unsigned* ua = (unsigned*)addr;
    unsigned old = *ua;
    while (true) {
        float f = __uint_as_float(old);
        if (f >= val) break;
        unsigned assumed = old;
        old = atomicCAS(ua, assumed, __float_as_uint(val));
        if (old == assumed) break;
    }
}

// order-preserving float<->uint key
__device__ __forceinline__ unsigned fkey(float f) {
    unsigned u = __float_as_uint(f);
    return (u & 0x80000000u) ? ~u : (u | 0x80000000u);
}
__device__ __forceinline__ float keyToFloat(unsigned k) {
    unsigned bits = (k & 0x80000000u) ? (k & 0x7FFFFFFFu) : ~k;
    return __uint_as_float(bits);
}

// ---------------- 0: fp32 -> bf16 convert ----------------
__global__ void cvt_bf16(const float* __restrict__ src, __bf16* __restrict__ dst, int n) {
    int t = blockIdx.x * 256 + threadIdx.x;
    if (t < n) dst[t] = (__bf16)src[t];
}

// ---------------- 1: QKV projection GEMM (WMMA + async-LDS double buffer) ----------------
// Out = X @ W^T + b, written in heads layout [bh][L][HD] as bf16.
// block tile: M=128 (grid.y), N=64 (grid.x), K stepped by 32. 8 waves, each 32x32.
__global__ __launch_bounds__(256) void qkv_gemm(
    const __bf16* __restrict__ Xb,
    const __bf16* __restrict__ Wq, const __bf16* __restrict__ Wk, const __bf16* __restrict__ Wv,
    const float* __restrict__ bq, const float* __restrict__ bk, const float* __restrict__ bv,
    __bf16* __restrict__ Qo, __bf16* __restrict__ Ko, __bf16* __restrict__ Vo)
{
    const __bf16* W  = (blockIdx.z == 0) ? Wq : (blockIdx.z == 1) ? Wk : Wv;
    const float* bias = (blockIdx.z == 0) ? bq : (blockIdx.z == 1) ? bk : bv;
    __bf16* Out       = (blockIdx.z == 0) ? Qo : (blockIdx.z == 1) ? Ko : Vo;

    __shared__ __bf16 As[2][128][40];   // row stride 80B (16B aligned), double buffered
    __shared__ __bf16 Bs[2][64][40];

    const int tid  = threadIdx.x;
    const int lane = tid & 31;
    const int wave = tid >> 5;
    const int wm = wave & 3;   // 4 M-subtiles of 32
    const int wn = wave >> 2;  // 2 N-subtiles of 32
    const int half = (lane >> 4) & 1;
    const int l16  = lane & 15;

    v8f zero = {0.f,0.f,0.f,0.f,0.f,0.f,0.f,0.f};
    v8f acc[2][2];
#pragma unroll
    for (int a = 0; a < 2; ++a)
#pragma unroll
        for (int b = 0; b < 2; ++b) acc[a][b] = zero;

    const int arow = tid >> 1, aseg = (tid & 1) << 4;
    const int brow = tid >> 2, bseg = (tid & 3) << 3;
    const size_t gArowOff = (size_t)(blockIdx.y * 128 + arow) * DM;
    const size_t gBrowOff = (size_t)(blockIdx.x * 64 + brow) * DM;

    // issue async loads of one K-tile into buffer `buf`: 3 instructions per wave
    auto issueTile = [&](int kb, int buf) {
        const __bf16* gA = Xb + gArowOff + kb + aseg;
        async_copy_b128(&As[buf][arow][aseg],     gA);
        async_copy_b128(&As[buf][arow][aseg + 8], gA + 8);
        async_copy_b128(&Bs[buf][brow][bseg],     W + gBrowOff + kb + bseg);
    };

    issueTile(0, 0);

    int p = 0;
    for (int kb = 0; kb < DM; kb += 32, p ^= 1) {
        if (kb + 32 < DM) {
            issueTile(kb + 32, p ^ 1);                       // overlap next-tile DMA with compute
            asm volatile("s_wait_asynccnt 3" ::: "memory");  // oldest 3 (this tile) complete
        } else {
            asm volatile("s_wait_asynccnt 0" ::: "memory");
        }
        __syncthreads();

        v16bf afrag[2], bfrag[2];
#pragma unroll
        for (int tm = 0; tm < 2; ++tm) {
            const __bf16* q = &As[p][wm * 32 + tm * 16 + l16][0];
            afrag[tm] = pack16(*(const v8bf*)(q + half * 16), *(const v8bf*)(q + half * 16 + 8));
        }
#pragma unroll
        for (int tn = 0; tn < 2; ++tn) {
            const __bf16* q = &Bs[p][wn * 32 + tn * 16 + l16][half * 16];
            bfrag[tn] = pack16(*(const v8bf*)q, *(const v8bf*)(q + 8));
        }
#pragma unroll
        for (int tm = 0; tm < 2; ++tm)
#pragma unroll
            for (int tn = 0; tn < 2; ++tn)
                acc[tm][tn] = wmma_bf16(afrag[tm], bfrag[tn], acc[tm][tn]);
        __syncthreads();   // all reads of buffer p done before it is overwritten next iter
    }

#pragma unroll
    for (int tm = 0; tm < 2; ++tm)
#pragma unroll
        for (int tn = 0; tn < 2; ++tn) {
            int nn = blockIdx.x * 64 + wn * 32 + tn * 16 + l16;
            float bv_ = bias[nn];
            int h = nn >> 6, d0 = nn & 63;
#pragma unroll
            for (int r = 0; r < 8; ++r) {
                int m = blockIdx.y * 128 + wm * 32 + tm * 16 + half * 8 + r;
                int b = m >> 12, l = m & 4095;
                Out[((size_t)(b * NHEADS + h) * SEQ + l) * HD + d0] = (__bf16)(acc[tm][tn][r] + bv_);
            }
        }
}

// ---------------- 2: block means ----------------
__global__ __launch_bounds__(64) void block_means(
    const __bf16* __restrict__ Qb, const __bf16* __restrict__ Kb, const __bf16* __restrict__ Vb,
    float* __restrict__ Qh, float* __restrict__ Kh, float* __restrict__ Vh)
{
    const __bf16* src = (blockIdx.y == 0) ? Qb : (blockIdx.y == 1) ? Kb : Vb;
    float* dst        = (blockIdx.y == 0) ? Qh : (blockIdx.y == 1) ? Kh : Vh;
    int bi = blockIdx.x;         // bh*NBLK + i
    int d0 = threadIdx.x;
    const __bf16* p = src + (size_t)bi * (BLK * HD) + d0;
    float s = 0.f;
#pragma unroll
    for (int r = 0; r < BLK; ++r) s += (float)p[r * HD];
    dst[(size_t)bi * HD + d0] = s * (1.f / BLK);
}

// ---------------- 3: low-res scores ----------------
__global__ __launch_bounds__(128) void low_scores(
    const float* __restrict__ Qh, const float* __restrict__ Kh,
    float* __restrict__ normb, float* __restrict__ rowmax)
{
    int bh = blockIdx.x, i = threadIdx.x;
    __shared__ float Ks[NBLK][HD];
    for (int t = threadIdx.x; t < NBLK * HD; t += 128) Ks[t >> 6][t & 63] = Kh[(size_t)bh * NBLK * HD + t];
    __syncthreads();

    float q[HD];
    const float* qp = Qh + (size_t)bh * NBLK * HD + (size_t)i * HD;
#pragma unroll
    for (int d = 0; d < HD; ++d) q[d] = qp[d];

    float* nrow = normb + (size_t)bh * NBLK * NBLK + (size_t)i * NBLK;
    float rmax = -3.4e38f;
    for (int j = 0; j < NBLK; ++j) {
        float s = 0.f;
#pragma unroll
        for (int d = 0; d < HD; ++d) s += q[d] * Ks[j][d];
        s *= SCALE;
        nrow[j] = s;
        rmax = fmaxf(rmax, s);
    }
    rowmax[bh * NBLK + i] = rmax;
    for (int j = 0; j < NBLK; ++j) {
        int dd = i - j; if (dd < 0) dd = -dd;
        nrow[j] = nrow[j] - rmax + ((dd <= 1) ? 5000.f : 0.f);
    }
}

// ---------------- 4: exact k-th threshold + gather ----------------
__global__ __launch_bounds__(256) void topk_select(
    const float* __restrict__ normb, float* __restrict__ thresh,
    int* __restrict__ idxb, int* __restrict__ cnt)
{
    int bh = blockIdx.x;
    const float* src = normb + (size_t)bh * NBLK * NBLK;
    __shared__ int red[256];

    unsigned lo = 0u, hi = 0xFFFFFFFFu;
    while (lo < hi) {
        unsigned long long span = (unsigned long long)hi - lo + 1ull;
        unsigned mid = lo + (unsigned)(span >> 1);
        int c = 0;
        for (int t = threadIdx.x; t < NBLK * NBLK; t += 256) c += (fkey(src[t]) >= mid) ? 1 : 0;
        red[threadIdx.x] = c;
        __syncthreads();
        for (int s = 128; s; s >>= 1) {
            if (threadIdx.x < s) red[threadIdx.x] += red[threadIdx.x + s];
            __syncthreads();
        }
        int total = red[0];
        __syncthreads();
        if (total >= NUMSEL) lo = mid; else hi = mid - 1;
    }
    if (threadIdx.x == 0) thresh[bh] = keyToFloat(lo);
    for (int t = threadIdx.x; t < NBLK * NBLK; t += 256) {
        if (fkey(src[t]) >= lo) {
            int p = atomicAdd(&cnt[bh], 1);
            if (p < NUMSEL) idxb[bh * NUMSEL + p] = t;
        }
    }
}

// ---------------- 5: init max_vals ----------------
__global__ void init_maxv(float* __restrict__ maxv) {
    int t = blockIdx.x * 256 + threadIdx.x;
    if (t < BH * NBLK * BLK) maxv[t] = -1e9f;
}

// ---------------- 6a: high-res row max (WMMA QK^T) ----------------
__global__ __launch_bounds__(32) void hi_rowmax(
    const __bf16* __restrict__ Qb, const __bf16* __restrict__ Kb,
    const int* __restrict__ idxb, float* __restrict__ maxv)
{
    int g = blockIdx.x, bh = g >> 10, slot = g & (NUMSEL - 1);
    int id = idxb[bh * NUMSEL + slot];
    int qblk = id >> 7, kblk = id & (NBLK - 1);
    int lane = threadIdx.x, half = lane >> 4, l16 = lane & 15;

    const __bf16* Qbase = Qb + (size_t)bh * SEQ * HD + (size_t)qblk * BLK * HD;
    const __bf16* Kbase = Kb + (size_t)bh * SEQ * HD + (size_t)kblk * BLK * HD;

    v8f zero = {0.f,0.f,0.f,0.f,0.f,0.f,0.f,0.f};
    v8f s[2][2] = {{zero, zero}, {zero, zero}};

#pragma unroll
    for (int kk = 0; kk < 2; ++kk) {
        v16bf a[2], b[2];
#pragma unroll
        for (int tm = 0; tm < 2; ++tm) {
            const __bf16* p = Qbase + (size_t)(tm * 16 + l16) * HD + kk * 32;
            a[tm] = pack16(*(const v8bf*)(p + half * 16), *(const v8bf*)(p + half * 16 + 8));
        }
#pragma unroll
        for (int tn = 0; tn < 2; ++tn) {
            const __bf16* p = Kbase + (size_t)(tn * 16 + l16) * HD + kk * 32 + half * 16;
            b[tn] = pack16(*(const v8bf*)p, *(const v8bf*)(p + 8));
        }
#pragma unroll
        for (int tm = 0; tm < 2; ++tm)
#pragma unroll
            for (int tn = 0; tn < 2; ++tn) s[tm][tn] = wmma_bf16(a[tm], b[tn], s[tm][tn]);
    }

#pragma unroll
    for (int tm = 0; tm < 2; ++tm)
#pragma unroll
        for (int r = 0; r < 8; ++r) {
            float v = fmaxf(s[tm][0][r], s[tm][1][r]) * SCALE;
#pragma unroll
            for (int m = 8; m >= 1; m >>= 1) v = fmaxf(v, __shfl_xor(v, m, 16));
            if (l16 == 0)
                atomicMaxF(&maxv[bh * (NBLK * BLK) + qblk * BLK + tm * 16 + half * 8 + r], v);
        }
}

// ---------------- 6b: high-res attention (WMMA QK^T, exp, WMMA P.V) ----------------
__global__ __launch_bounds__(32) void hi_attn(
    const __bf16* __restrict__ Qb, const __bf16* __restrict__ Kb, const __bf16* __restrict__ Vb,
    const int* __restrict__ idxb, const float* __restrict__ maxv,
    float* __restrict__ hi_out, float* __restrict__ hi_norm)
{
    int g = blockIdx.x, bh = g >> 10, slot = g & (NUMSEL - 1);
    int id = idxb[bh * NUMSEL + slot];
    int qblk = id >> 7, kblk = id & (NBLK - 1);
    int lane = threadIdx.x, half = lane >> 4, l16 = lane & 15;

    const __bf16* Qbase = Qb + (size_t)bh * SEQ * HD + (size_t)qblk * BLK * HD;
    const __bf16* Kbase = Kb + (size_t)bh * SEQ * HD + (size_t)kblk * BLK * HD;
    const __bf16* Vbase = Vb + (size_t)bh * SEQ * HD + (size_t)kblk * BLK * HD;

    __shared__ float  mv[BLK];
    __shared__ __bf16 Pst[32][40];
    __shared__ __bf16 Vt[64][40];

    mv[lane] = maxv[bh * (NBLK * BLK) + qblk * BLK + lane];
    // stage V transposed: lane = key row k, write column
    {
        const __bf16* vp = Vbase + (size_t)lane * HD;
#pragma unroll
        for (int d = 0; d < HD; ++d) Vt[d][lane] = vp[d];
    }

    v8f zero = {0.f,0.f,0.f,0.f,0.f,0.f,0.f,0.f};
    v8f s[2][2] = {{zero, zero}, {zero, zero}};
#pragma unroll
    for (int kk = 0; kk < 2; ++kk) {
        v16bf a[2], b[2];
#pragma unroll
        for (int tm = 0; tm < 2; ++tm) {
            const __bf16* p = Qbase + (size_t)(tm * 16 + l16) * HD + kk * 32;
            a[tm] = pack16(*(const v8bf*)(p + half * 16), *(const v8bf*)(p + half * 16 + 8));
        }
#pragma unroll
        for (int tn = 0; tn < 2; ++tn) {
            const __bf16* p = Kbase + (size_t)(tn * 16 + l16) * HD + kk * 32 + half * 16;
            b[tn] = pack16(*(const v8bf*)p, *(const v8bf*)(p + 8));
        }
#pragma unroll
        for (int tm = 0; tm < 2; ++tm)
#pragma unroll
            for (int tn = 0; tn < 2; ++tn) s[tm][tn] = wmma_bf16(a[tm], b[tn], s[tm][tn]);
    }
    __syncthreads();   // mv + Vt visible

    // P = exp(S*SCALE - max), row sums -> hi_norm, bf16 P -> LDS
#pragma unroll
    for (int tm = 0; tm < 2; ++tm) {
#pragma unroll
        for (int r = 0; r < 8; ++r) {
            int row = tm * 16 + half * 8 + r;
            float m0 = mv[row];
            float p0 = __expf(s[tm][0][r] * SCALE - m0);
            float p1 = __expf(s[tm][1][r] * SCALE - m0);
            Pst[row][l16]      = (__bf16)p0;
            Pst[row][16 + l16] = (__bf16)p1;
            float rs = p0 + p1;
#pragma unroll
            for (int m = 8; m >= 1; m >>= 1) rs += __shfl_xor(rs, m, 16);
            if (l16 == 0) atomicAdd(&hi_norm[(size_t)bh * SEQ + qblk * BLK + row], rs);
        }
    }
    __syncthreads();

    // O = P(32x32) @ V(32x64)  (K=32, one WMMA per 16x16 tile)
    v16bf pa[2];
#pragma unroll
    for (int tm = 0; tm < 2; ++tm) {
        const __bf16* p = &Pst[tm * 16 + l16][0];
        pa[tm] = pack16(*(const v8bf*)(p + half * 16), *(const v8bf*)(p + half * 16 + 8));
    }
#pragma unroll
    for (int tnn = 0; tnn < 4; ++tnn) {
        const __bf16* p = &Vt[tnn * 16 + l16][half * 16];
        v16bf vb = pack16(*(const v8bf*)p, *(const v8bf*)(p + 8));
#pragma unroll
        for (int tm = 0; tm < 2; ++tm) {
            v8f o = wmma_bf16(pa[tm], vb, zero);
#pragma unroll
            for (int r = 0; r < 8; ++r) {
                int row = qblk * BLK + tm * 16 + half * 8 + r;
                atomicAdd(&hi_out[((size_t)bh * SEQ + row) * HD + tnn * 16 + l16], o[r]);
            }
        }
    }
}

// ---------------- 7: low-res path ----------------
__global__ __launch_bounds__(128) void low_path(
    const float* __restrict__ normb, const float* __restrict__ thresh,
    const float* __restrict__ Vh, float* __restrict__ low_out, float* __restrict__ low_norm)
{
    int bh = blockIdx.x, i = threadIdx.x;
    __shared__ float Vs[NBLK][HD];
    for (int t = threadIdx.x; t < NBLK * HD; t += 128) Vs[t >> 6][t & 63] = Vh[(size_t)bh * NBLK * HD + t];
    __syncthreads();

    const float* nrow = normb + (size_t)bh * NBLK * NBLK + (size_t)i * NBLK;
    float th = thresh[bh];
    float acc[HD];
#pragma unroll
    for (int d = 0; d < HD; ++d) acc[d] = 0.f;
    float wsum = 0.f;
    for (int j = 0; j < NBLK; ++j) {
        float nv = nrow[j];
        int dd = i - j; if (dd < 0) dd = -dd;
        float e = nv - ((dd <= 1) ? 5000.f : 0.f) - ((nv >= th) ? 10000.f : 0.f);
        float w = __expf(e) * (float)BLK;
        wsum += w;
#pragma unroll
        for (int d = 0; d < HD; ++d) acc[d] += w * Vs[j][d];
    }
#pragma unroll
    for (int d = 0; d < HD; ++d) low_out[((size_t)bh * NBLK + i) * HD + d] = acc[d];
    low_norm[bh * NBLK + i] = wsum;
}

// ---------------- 8: combine + head-merge output ----------------
__global__ void combine(
    const float* __restrict__ hi_out, const float* __restrict__ hi_norm,
    const float* __restrict__ low_out, const float* __restrict__ low_norm,
    const float* __restrict__ rowmax, const float* __restrict__ maxv,
    float* __restrict__ out)
{
    size_t t = (size_t)blockIdx.x * 256 + threadIdx.x;
    if (t >= (size_t)BATCH * SEQ * DM) return;
    int dcol = (int)(t & (DM - 1));
    size_t bl = t >> 10;
    int l = (int)(bl & (SEQ - 1));
    int b = (int)(bl >> 12);
    int h = dcol >> 6, d0 = dcol & 63;
    int bh = b * NHEADS + h;
    int i = l >> 5, r = l & 31;

    float rm = rowmax[bh * NBLK + i];
    float mv = maxv[bh * (NBLK * BLK) + i * BLK + r];
    float lc = rm - mv;
    float low_corr = __expf(fminf(lc, 0.f));
    float hi_corr  = __expf(-fmaxf(lc, 0.f));
    float ho = hi_out[((size_t)bh * SEQ + l) * HD + d0];
    float lo = low_out[((size_t)bh * NBLK + i) * HD + d0];
    float hn = hi_norm[(size_t)bh * SEQ + l];
    float ln = low_norm[bh * NBLK + i];
    float num = ho * hi_corr + lo * low_corr;
    float den = hn * hi_corr + ln * low_corr + 1e-6f;
    out[t] = num / den;
}

// ---------------- host launcher ----------------
extern "C" void kernel_launch(void* const* d_in, const int* in_sizes, int n_in,
                              void* d_out, int out_size, void* d_ws, size_t ws_size,
                              hipStream_t stream) {
    (void)in_sizes; (void)n_in; (void)out_size; (void)ws_size;
    const float* X  = (const float*)d_in[0];
    const float* Wq = (const float*)d_in[1];
    const float* bq = (const float*)d_in[2];
    const float* Wk = (const float*)d_in[3];
    const float* bk = (const float*)d_in[4];
    const float* Wv = (const float*)d_in[5];
    const float* bv = (const float*)d_in[6];
    float* out = (float*)d_out;

    const size_t NX  = (size_t)BATCH * SEQ * DM;   // 8388608
    const size_t ND2 = (size_t)DM * DM;            // 1048576

    char* ws = (char*)d_ws;
    size_t off = 0;
    auto alloc = [&](size_t bytes) -> char* {
        char* p = ws + off;
        off = (off + bytes + 255) & ~(size_t)255;
        return p;
    };
    __bf16* Xb  = (__bf16*)alloc(NX * 2);
    __bf16* Wqb = (__bf16*)alloc(ND2 * 2);
    __bf16* Wkb = (__bf16*)alloc(ND2 * 2);
    __bf16* Wvb = (__bf16*)alloc(ND2 * 2);
    __bf16* Qb  = (__bf16*)alloc(NX * 2);
    __bf16* Kb  = (__bf16*)alloc(NX * 2);
    __bf16* Vb  = (__bf16*)alloc(NX * 2);
    float* Qh   = (float*)alloc((size_t)BH * NBLK * HD * 4);
    float* Kh   = (float*)alloc((size_t)BH * NBLK * HD * 4);
    float* Vh   = (float*)alloc((size_t)BH * NBLK * HD * 4);
    float* normb   = (float*)alloc((size_t)BH * NBLK * NBLK * 4);
    float* rowmax  = (float*)alloc((size_t)BH * NBLK * 4);
    float* thresh  = (float*)alloc((size_t)BH * 4);
    int*   cnt     = (int*)  alloc((size_t)BH * 4);
    int*   idxb    = (int*)  alloc((size_t)BH * NUMSEL * 4);
    float* maxv    = (float*)alloc((size_t)BH * NBLK * BLK * 4);
    float* hi_outb = (float*)alloc(NX * 4);
    float* hi_normb= (float*)alloc((size_t)BH * SEQ * 4);
    float* low_outb= (float*)alloc((size_t)BH * NBLK * HD * 4);
    float* low_normb=(float*)alloc((size_t)BH * NBLK * 4);

    // conversions
    cvt_bf16<<<(int)(NX / 256), 256, 0, stream>>>(X, Xb, (int)NX);
    cvt_bf16<<<(int)(ND2 / 256), 256, 0, stream>>>(Wq, Wqb, (int)ND2);
    cvt_bf16<<<(int)(ND2 / 256), 256, 0, stream>>>(Wk, Wkb, (int)ND2);
    cvt_bf16<<<(int)(ND2 / 256), 256, 0, stream>>>(Wv, Wvb, (int)ND2);

    // QKV projections (WMMA + async-LDS pipeline)
    qkv_gemm<<<dim3(DM / 64, (BATCH * SEQ) / 128, 3), 256, 0, stream>>>(
        Xb, Wqb, Wkb, Wvb, bq, bk, bv, Qb, Kb, Vb);

    // block means
    block_means<<<dim3(BH * NBLK, 3), 64, 0, stream>>>(Qb, Kb, Vb, Qh, Kh, Vh);

    // low-res scores
    low_scores<<<BH, 128, 0, stream>>>(Qh, Kh, normb, rowmax);

    // top-k selection
    hipMemsetAsync(cnt, 0, (size_t)BH * 4, stream);
    topk_select<<<BH, 256, 0, stream>>>(normb, thresh, idxb, cnt);

    // init accumulators
    hipMemsetAsync(hi_outb, 0, NX * 4, stream);
    hipMemsetAsync(hi_normb, 0, (size_t)BH * SEQ * 4, stream);
    init_maxv<<<(BH * NBLK * BLK) / 256, 256, 0, stream>>>(maxv);

    // high-res attention
    hi_rowmax<<<BH * NUMSEL, 32, 0, stream>>>(Qb, Kb, idxb, maxv);
    hi_attn<<<BH * NUMSEL, 32, 0, stream>>>(Qb, Kb, Vb, idxb, maxv, hi_outb, hi_normb);

    // low-res path
    low_path<<<BH, 128, 0, stream>>>(normb, thresh, Vh, low_outb, low_normb);

    // combine
    combine<<<(int)(NX / 256), 256, 0, stream>>>(
        hi_outb, hi_normb, low_outb, low_normb, rowmax, maxv, out);
}
